// GRUClassifier_69844758168141
// MI455X (gfx1250) — compile-verified, matrix operand order checked by reading, not run on hardware
//
#include <hip/hip_runtime.h>
#include <hip/hip_bf16.h>

#define NH_   1024
#define CATS_ 1024
#define B_    64
#define T_    512
#define NWG_SCAN 16

typedef __attribute__((ext_vector_type(16))) __bf16 v16bf;
typedef __attribute__((ext_vector_type(8)))  __bf16 v8bf;
typedef __attribute__((ext_vector_type(8)))  float  v8f;
typedef __attribute__((ext_vector_type(4)))  float  v4f;
typedef __attribute__((ext_vector_type(4)))  int    v4i;

#define WMMA_BF16(a, b, c) \
  __builtin_amdgcn_wmma_f32_16x16x32_bf16(false, (a), false, (b), (short)0, (c), false, false)

// ---- CDNA5 async global->LDS copy (guarded; falls back to reg-staged copy) --
#if defined(__has_builtin)
#if __has_builtin(__builtin_amdgcn_global_load_async_to_lds_b128)
#define HAS_ASYNC_LDS 1
#endif
#endif
#ifndef HAS_ASYNC_LDS
#define HAS_ASYNC_LDS 0
#endif

typedef __attribute__((address_space(1))) v4i gv4i;  // global int4
typedef __attribute__((address_space(3))) v4i lv4i;  // LDS int4

__device__ __forceinline__ void copy16_g2l(const __bf16* gsrc, __bf16* ldst) {
#if HAS_ASYNC_LDS
  __builtin_amdgcn_global_load_async_to_lds_b128((gv4i*)gsrc, (lv4i*)ldst, 0, 0);
#else
  *(v8bf*)ldst = *(const v8bf*)gsrc;
#endif
}

__device__ __forceinline__ void wait_async_copies() {
#if HAS_ASYNC_LDS
#if __has_builtin(__builtin_amdgcn_s_wait_asynccnt)
  __builtin_amdgcn_s_wait_asynccnt(0);
#else
  asm volatile("s_wait_asynccnt 0x0" ::: "memory");
#endif
#endif
}

__device__ __forceinline__ v8bf cvt8(v4f a, v4f b) {
  v8bf r;
  r[0] = (__bf16)a[0]; r[1] = (__bf16)a[1]; r[2] = (__bf16)a[2]; r[3] = (__bf16)a[3];
  r[4] = (__bf16)b[0]; r[5] = (__bf16)b[1]; r[6] = (__bf16)b[2]; r[7] = (__bf16)b[3];
  return r;
}

__device__ __forceinline__ v16bf merge16(v8bf a, v8bf b) {
  v16bf f;
#pragma unroll
  for (int i = 0; i < 8; ++i) { f[i] = a[i]; f[8 + i] = b[i]; }
  return f;
}

// ISA 16-bit fragment layout: lane row = lane&15; lane<16 holds k in
// {0..7, 16..23}+k0, lane>=16 holds k in {8..15, 24..31}+k0.
__device__ __forceinline__ v16bf frag_from_f32(const float* p, int hi) {
  const v4f* q0 = (const v4f*)(p + hi * 8);
  const v4f* q1 = (const v4f*)(p + hi * 8 + 16);
  return merge16(cvt8(q0[0], q0[1]), cvt8(q1[0], q1[1]));
}

__device__ __forceinline__ v16bf frag_from_bf16(const __bf16* p, int hi) {
  v8bf a = *(const v8bf*)(p + hi * 8);
  v8bf b = *(const v8bf*)(p + hi * 8 + 16);
  return merge16(a, b);
}

// ---------------------------------------------------------------------------
// Prep: f32 -> bf16 elementwise conversion
// ---------------------------------------------------------------------------
__global__ void cvt_f32_bf16_kernel(const float* __restrict__ src,
                                    __bf16* __restrict__ dst, int n) {
  int i = blockIdx.x * blockDim.x + threadIdx.x;
  if (i < n) dst[i] = (__bf16)src[i];
}

// ---------------------------------------------------------------------------
// Phase 1: xh = x @ Wh^T, xz = x @ Wz^T   (M = B*T = 32768, K = N = 1024)
// Output layout: [T][B][NH] so the scan reads one contiguous [B][NH] per step.
// Workgroup: 8 waves; WG tile = 128(M) x 64(N). Pre-converted bf16 Wh/Wz
// k-slices are double-buffered in LDS via async global->LDS copies so the
// tile(k+1) copy overlaps the tile(k) WMMAs; one wait+barrier per iteration.
// ---------------------------------------------------------------------------
__global__ __launch_bounds__(256) void xproj_kernel(
    const float* __restrict__ x,     // [B*T, CATS] f32
    const __bf16* __restrict__ Whb,  // [NH, CATS] bf16
    const __bf16* __restrict__ Wzb,  // [NH, CATS] bf16
    float* __restrict__ xh,          // [T*B, NH]
    float* __restrict__ xz) {        // [T*B, NH]
  __shared__ __align__(16) __bf16 ldsW[2][2][64][32];  // [buf][gate][n][k] 16KB

  const int tid   = threadIdx.x;
  const int wave  = tid >> 5;
  const int lane  = tid & 31;
  const int row16 = lane & 15;
  const int hi    = lane >> 4;
  const int mBase = blockIdx.x * 128 + wave * 16;
  const int nBase = blockIdx.y * 64;

  // Per-thread staging slot: 32 bytes of the 8KB [gate][n][k] tile.
  const int off  = tid * 32;           // byte offset within tile
  const int gate = off >> 12;          // 0: Wh, 1: Wz
  const int n_st = (off & 4095) >> 6;  // row 0..63
  const int kel  = (off & 63) >> 1;    // element offset within row (0 or 16)
  const __bf16* wsrc = (gate ? Wzb : Whb) + (size_t)(nBase + n_st) * CATS_ + kel;
  __bf16* ldst[2] = { &ldsW[0][gate][n_st][kel], &ldsW[1][gate][n_st][kel] };

  v8f acc[2][4] = {};
  const float* xrow = x + (size_t)(mBase + row16) * CATS_;

  // Prologue: stage tile k0=0 into buffer 0.
  copy16_g2l(wsrc, ldst[0]);
  copy16_g2l(wsrc + 8, ldst[0] + 8);
  wait_async_copies();
  __syncthreads();

  int p = 0;
  for (int k0 = 0; k0 < CATS_; k0 += 32) {
    if (k0 + 32 < CATS_) {  // kick off next tile copy into the other buffer
      copy16_g2l(wsrc + k0 + 32, ldst[p ^ 1]);
      copy16_g2l(wsrc + k0 + 40, ldst[p ^ 1] + 8);
      __builtin_prefetch(xrow + k0 + 32, 0, 1);
    }

    v16bf afrag = frag_from_f32(xrow + k0, hi);

    // Load all B fragments first -> one dscnt wait, then 8 chained WMMAs.
    v16bf bfr[8];
#pragma unroll
    for (int nt = 0; nt < 4; ++nt) {
      int nl = nt * 16 + row16;
      bfr[nt]     = merge16(*(const v8bf*)&ldsW[p][0][nl][hi * 8],
                            *(const v8bf*)&ldsW[p][0][nl][hi * 8 + 16]);
      bfr[4 + nt] = merge16(*(const v8bf*)&ldsW[p][1][nl][hi * 8],
                            *(const v8bf*)&ldsW[p][1][nl][hi * 8 + 16]);
    }
#pragma unroll
    for (int nt = 0; nt < 4; ++nt) {
      acc[0][nt] = WMMA_BF16(afrag, bfr[nt],     acc[0][nt]);
      acc[1][nt] = WMMA_BF16(afrag, bfr[4 + nt], acc[1][nt]);
    }

    wait_async_copies();
    __syncthreads();
    p ^= 1;
  }

  // C layout: element r -> M = mBase + r + 8*hi, N = nBase + nt*16 + (lane&15)
#pragma unroll
  for (int nt = 0; nt < 4; ++nt) {
    int n = nBase + nt * 16 + row16;
#pragma unroll
    for (int r = 0; r < 8; ++r) {
      int m = mBase + r + 8 * hi;  // m = b*T + t
      int b = m >> 9;              // / T_
      int t = m & (T_ - 1);
      size_t o = ((size_t)(t * B_ + b)) * NH_ + n;
      xh[o] = acc[0][nt][r];
      xz[o] = acc[1][nt][r];
    }
  }
}

// ---------------------------------------------------------------------------
// Phase 2: persistent GRU scan. 16 WGs, WG g owns columns [g*64, g*64+64).
// Wave w owns mTile = w>>1 and two n-tiles; keeps its h columns in f32 regs.
// Cross-WG h exchange via double-buffered bf16 array + global atomic barrier.
// ---------------------------------------------------------------------------
__global__ __launch_bounds__(256) void scan_kernel(
    const float* __restrict__ xh,    // [T*B, NH]
    const float* __restrict__ xz,    // [T*B, NH]
    const float* __restrict__ h0,    // [B, NH]
    const __bf16* __restrict__ Uh,   // [NH, NH] bf16
    const __bf16* __restrict__ Uz,   // [NH, NH] bf16
    const float* __restrict__ bz,    // [NH]
    __bf16* __restrict__ hbuf,       // [2][B, NH] bf16 (buf0 preloaded = h0)
    unsigned* __restrict__ ctr) {
  const int tid      = threadIdx.x;
  const int wave     = tid >> 5;
  const int lane     = tid & 31;
  const int row16    = lane & 15;
  const int hi       = lane >> 4;
  const int g        = blockIdx.x;      // 0..15
  const int mTile    = wave >> 1;       // 0..3
  const int ntBase   = (wave & 1) * 2;  // {0,2}
  const int nColBase = g * 64;

  v8f   hreg[2];
  float bzv[2];
#pragma unroll
  for (int j = 0; j < 2; ++j) {
    int n  = nColBase + (ntBase + j) * 16 + row16;
    bzv[j] = bz[n];
#pragma unroll
    for (int r = 0; r < 8; ++r) {
      int b = mTile * 16 + r + 8 * hi;
      hreg[j][r] = h0[(size_t)b * NH_ + n];
    }
  }

  const __bf16* urow[4];
#pragma unroll
  for (int j = 0; j < 2; ++j) {
    int n = nColBase + (ntBase + j) * 16 + row16;
    urow[j]     = Uz + (size_t)n * NH_;
    urow[2 + j] = Uh + (size_t)n * NH_;
  }

  for (int t = 0; t < T_; ++t) {
    const __bf16* hcur  = hbuf + (size_t)(t & 1) * (B_ * NH_);
    __bf16*       hnext = hbuf + (size_t)((t + 1) & 1) * (B_ * NH_);

    v8f az[2] = {};
    v8f ah[2] = {};
    const __bf16* arow = hcur + (size_t)(mTile * 16 + row16) * NH_;

#pragma unroll 4
    for (int k0 = 0; k0 < NH_; k0 += 32) {
      v16bf afrag = frag_from_bf16(arow + k0, hi);
      v16bf bfr[4];
#pragma unroll
      for (int q = 0; q < 4; ++q) bfr[q] = frag_from_bf16(urow[q] + k0, hi);
#pragma unroll
      for (int j = 0; j < 2; ++j) {
        az[j] = WMMA_BF16(afrag, bfr[j],     az[j]);
        ah[j] = WMMA_BF16(afrag, bfr[2 + j], ah[j]);
      }
    }

    // Gate math + state update (f32).
#pragma unroll
    for (int j = 0; j < 2; ++j) {
      int n = nColBase + (ntBase + j) * 16 + row16;
#pragma unroll
      for (int r = 0; r < 8; ++r) {
        int b = mTile * 16 + r + 8 * hi;
        size_t xo = ((size_t)(t * B_ + b)) * NH_ + n;
        float zi = az[j][r] + bzv[j] + xz[xo];
        float z  = 1.0f / (1.0f + __expf(-zi));
        float pi = ah[j][r] + xh[xo];
        float e  = __expf(-2.0f * pi);
        float ht = (1.0f - e) / (1.0f + e);  // tanh
        float hn = (1.0f - z) * hreg[j][r] + z * ht;
        hreg[j][r] = hn;
        hnext[(size_t)b * NH_ + n] = (__bf16)hn;
      }
    }

    // Grid-wide step barrier (monotonic counter, reset per launch).
    __threadfence();
    __syncthreads();
    if (tid == 0) {
      atomicAdd(ctr, 1u);
      unsigned target = (unsigned)(t + 1) * NWG_SCAN;
      while (__hip_atomic_load(ctr, __ATOMIC_RELAXED, __HIP_MEMORY_SCOPE_AGENT) < target) {
        __builtin_amdgcn_s_sleep(2);
      }
    }
    __syncthreads();
    __threadfence();
  }
  // Final hT sits in hbuf[T_ & 1] == hbuf[0].
}

// ---------------------------------------------------------------------------
// Phase 3: out = hT @ Wout^T   (64 x 1024, K = 1024)
// ---------------------------------------------------------------------------
__global__ __launch_bounds__(256) void out_kernel(
    const __bf16* __restrict__ hT,    // [B, NH] bf16 (hbuf[0])
    const __bf16* __restrict__ Wout,  // [CATS, NH] bf16
    float* __restrict__ out) {        // [B, CATS]
  const int tid   = threadIdx.x;
  const int wave  = tid >> 5;
  const int lane  = tid & 31;
  const int row16 = lane & 15;
  const int hi    = lane >> 4;
  const int gid   = blockIdx.x * 8 + wave;  // 0..63
  const int mBase = (gid >> 4) * 16;
  const int nBase = (gid & 15) * 64;

  v8f acc[4] = {};
  const __bf16* arow = hT + (size_t)(mBase + row16) * NH_;

  for (int k0 = 0; k0 < NH_; k0 += 32) {
    v16bf afrag = frag_from_bf16(arow + k0, hi);
    v16bf bfr[4];
#pragma unroll
    for (int nt = 0; nt < 4; ++nt)
      bfr[nt] = frag_from_bf16(Wout + (size_t)(nBase + nt * 16 + row16) * NH_ + k0, hi);
#pragma unroll
    for (int nt = 0; nt < 4; ++nt)
      acc[nt] = WMMA_BF16(afrag, bfr[nt], acc[nt]);
  }

#pragma unroll
  for (int nt = 0; nt < 4; ++nt) {
    int n = nBase + nt * 16 + row16;
#pragma unroll
    for (int r = 0; r < 8; ++r) {
      int m = mBase + r + 8 * hi;
      out[(size_t)m * CATS_ + n] = acc[nt][r];
    }
  }
}

// ---------------------------------------------------------------------------
extern "C" void kernel_launch(void* const* d_in, const int* in_sizes, int n_in,
                              void* d_out, int out_size, void* d_ws, size_t ws_size,
                              hipStream_t stream) {
  (void)in_sizes; (void)n_in; (void)out_size; (void)ws_size;

  const float* x    = (const float*)d_in[0];
  const float* h0   = (const float*)d_in[1];
  const float* Wh   = (const float*)d_in[2];
  const float* Wz   = (const float*)d_in[3];
  // d_in[4] = Wr (dead in reference)
  const float* Uh   = (const float*)d_in[5];
  const float* Uz   = (const float*)d_in[6];
  // d_in[7] = Ur (dead)
  const float* bz   = (const float*)d_in[8];
  // d_in[9] = br (dead)
  const float* Wout = (const float*)d_in[10];
  float* out = (float*)d_out;

  // Workspace carve-up.
  char* ws = (char*)d_ws;
  const size_t S1 = (size_t)B_ * T_ * NH_ * sizeof(float);  // 134 MB each
  const size_t NW = (size_t)NH_ * NH_;
  float*  xh      = (float*)ws;
  float*  xz      = (float*)(ws + S1);
  __bf16* uh_bf   = (__bf16*)(ws + 2 * S1);
  __bf16* uz_bf   = uh_bf + NW;
  __bf16* wout_bf = uz_bf + NW;
  __bf16* wh_bf   = wout_bf + NW;
  __bf16* wz_bf   = wh_bf + NW;
  __bf16* hbuf    = wz_bf + NW;                  // 2 * [B,NH]
  unsigned* ctr   = (unsigned*)(hbuf + 2 * (size_t)B_ * NH_);

  (void)hipMemsetAsync(ctr, 0, sizeof(unsigned), stream);

  const int nW = NH_ * NH_;
  cvt_f32_bf16_kernel<<<(nW + 255) / 256, 256, 0, stream>>>(Uh, uh_bf, nW);
  cvt_f32_bf16_kernel<<<(nW + 255) / 256, 256, 0, stream>>>(Uz, uz_bf, nW);
  cvt_f32_bf16_kernel<<<(nW + 255) / 256, 256, 0, stream>>>(Wout, wout_bf, nW);
  cvt_f32_bf16_kernel<<<(nW + 255) / 256, 256, 0, stream>>>(Wh, wh_bf, nW);
  cvt_f32_bf16_kernel<<<(nW + 255) / 256, 256, 0, stream>>>(Wz, wz_bf, nW);
  cvt_f32_bf16_kernel<<<(B_ * NH_ + 255) / 256, 256, 0, stream>>>(h0, hbuf, B_ * NH_);

  dim3 g1(B_ * T_ / 128, NH_ / 64);  // 256 x 16 workgroups
  xproj_kernel<<<g1, 256, 0, stream>>>(x, wh_bf, wz_bf, xh, xz);

  scan_kernel<<<NWG_SCAN, 256, 0, stream>>>(xh, xz, h0, uh_bf, uz_bf, bz, hbuf, ctr);

  out_kernel<<<8, 256, 0, stream>>>(hbuf, wout_bf, out);
}
